// S4Block_18373870092479
// MI455X (gfx1250) — matrix operand order repaired and checked by reading, not compiled, not run on hardware
//
#include <hip/hip_runtime.h>

typedef __attribute__((ext_vector_type(16))) _Float16 v16h;
typedef __attribute__((ext_vector_type(8)))  _Float16 v8h;
typedef __attribute__((ext_vector_type(4)))  _Float16 v4h;
typedef __attribute__((ext_vector_type(8)))  float    v8f;

// Problem constants (fixed by the reference)
constexpr int BATCH = 4;
constexpr int SEQ   = 4096;
constexpr int DIM   = 1024;   // D
constexpr int NSTATE= 256;    // N
constexpr int MTOK  = BATCH * SEQ;  // 16384 tokens
constexpr int CHUNK = 256;          // scan chunk length
constexpr int NCH   = SEQ / CHUNK;  // 16 chunks per batch

// ---------------------------------------------------------------------------
// f32 -> (hi,lo) f16 split helpers
// ---------------------------------------------------------------------------
struct HL { _Float16 h, l; };

__device__ __forceinline__ HL split16(float x) {
  HL r;
  r.h = (_Float16)x;
  r.l = (_Float16)(x - (float)r.h);
  return r;
}

__device__ __forceinline__ v16h frag16(v8h a, v8h b) {
  return __builtin_shufflevector(a, b, 0,1,2,3,4,5,6,7,8,9,10,11,12,13,14,15);
}

// ---------------------------------------------------------------------------
// Kernel: generic f32 -> hi/lo f16 split (vectorized 4 elems/thread)
// ---------------------------------------------------------------------------
__global__ void k_convert_split(const float* __restrict__ src,
                                _Float16* __restrict__ hi,
                                _Float16* __restrict__ lo, int n4) {
  int i = blockIdx.x * blockDim.x + threadIdx.x;
  if (i >= n4) return;
  float4 v = ((const float4*)src)[i];
  HL a = split16(v.x), b = split16(v.y), c = split16(v.z), d = split16(v.w);
  v4h h, l;
  h[0] = a.h; h[1] = b.h; h[2] = c.h; h[3] = d.h;
  l[0] = a.l; l[1] = b.l; l[2] = c.l; l[3] = d.l;
  ((v4h*)hi)[i] = h;
  ((v4h*)lo)[i] = l;
}

// ---------------------------------------------------------------------------
// Kernel: RMSNorm (eps=1e-4) fused with hi/lo split.  1 block = 1 token row.
// ---------------------------------------------------------------------------
__global__ __launch_bounds__(256) void k_rmsnorm_split(
    const float* __restrict__ tokens, const float* __restrict__ norm_w,
    _Float16* __restrict__ xh, _Float16* __restrict__ xl) {
  const int row = blockIdx.x;
  const int t   = threadIdx.x;
  const size_t base = (size_t)row * DIM + t * 4;
  float4 v = *(const float4*)(tokens + base);
  float ss = v.x*v.x + v.y*v.y + v.z*v.z + v.w*v.w;
  // wave32 reduce
  #pragma unroll
  for (int o = 16; o > 0; o >>= 1) ss += __shfl_xor(ss, o, 32);
  __shared__ float part[8];
  if ((t & 31) == 0) part[t >> 5] = ss;
  __syncthreads();
  float tot = part[0]+part[1]+part[2]+part[3]+part[4]+part[5]+part[6]+part[7];
  float r = rsqrtf(tot * (1.0f / (float)DIM) + 1e-4f);
  float4 w = *(const float4*)(norm_w + t * 4);
  HL a = split16(v.x * r * w.x);
  HL b = split16(v.y * r * w.y);
  HL c = split16(v.z * r * w.z);
  HL d = split16(v.w * r * w.w);
  v4h h, l;
  h[0] = a.h; h[1] = b.h; h[2] = c.h; h[3] = d.h;
  l[0] = a.l; l[1] = b.l; l[2] = c.l; l[3] = d.l;
  *(v4h*)(xh + base) = h;
  *(v4h*)(xl + base) = l;
}

// ---------------------------------------------------------------------------
// GEMM mainloop: C[m,n] += A[m,:]·B[n,:] (NT, both K-contiguous), split-f16.
// Block tile 128(M) x 64(N) x 32(K); 256 threads = 8 waves (4x2 wave grid);
// each wave computes a 32x32 tile = 2x2 WMMA 16x16 tiles, 3 WMMAs each
// (hi*hi + hi*lo + lo*hi) with f32 accumulate.
// ---------------------------------------------------------------------------
__device__ __forceinline__ void gemm_mainloop(
    const _Float16* __restrict__ Agh, const _Float16* __restrict__ Agl, int lda,
    const _Float16* __restrict__ Bgh, const _Float16* __restrict__ Bgl, int ldb,
    int K, int m0, int n0,
    _Float16 (*sAh)[40], _Float16 (*sAl)[40],
    _Float16 (*sBh)[40], _Float16 (*sBl)[40],
    v8f acc[2][2]) {
  const int tid  = threadIdx.x;
  const int wave = tid >> 5, lane = tid & 31;
  const int wm = (wave >> 1) * 32, wn = (wave & 1) * 32;
  const int lr = lane & 15, lh = lane >> 4;

  for (int k0 = 0; k0 < K; k0 += 32) {
    // ---- global -> LDS: A tile 128x32 (hi & lo), 2 x 16B chunks per thread
    #pragma unroll
    for (int i = 0; i < 2; ++i) {
      int cid = tid + i * 256;
      int r = cid >> 2, kc = cid & 3;
      size_t g = (size_t)(m0 + r) * lda + (k0 + kc * 8);
      *(v8h*)&sAh[r][kc * 8] = *(const v8h*)(Agh + g);
      *(v8h*)&sAl[r][kc * 8] = *(const v8h*)(Agl + g);
    }
    { // ---- B tile 64x32 (hi & lo), 1 x 16B chunk per thread per array
      int r = tid >> 2, kc = tid & 3;
      size_t g = (size_t)(n0 + r) * ldb + (k0 + kc * 8);
      *(v8h*)&sBh[r][kc * 8] = *(const v8h*)(Bgh + g);
      *(v8h*)&sBl[r][kc * 8] = *(const v8h*)(Bgl + g);
    }
    __syncthreads();

    // ---- fragments (ISA §7.12.2 layouts)
    v16h aH[2], aL[2], bH[2], bL[2];
    #pragma unroll
    for (int mt = 0; mt < 2; ++mt) {
      const _Float16* p = &sAh[wm + mt * 16 + lr][0];
      const _Float16* q = &sAl[wm + mt * 16 + lr][0];
      aH[mt] = frag16(*(const v8h*)(p + 8 * lh), *(const v8h*)(p + 16 + 8 * lh));
      aL[mt] = frag16(*(const v8h*)(q + 8 * lh), *(const v8h*)(q + 16 + 8 * lh));
    }
    #pragma unroll
    for (int nt = 0; nt < 2; ++nt) {
      const _Float16* p = &sBh[wn + nt * 16 + lr][8 * lh];
      const _Float16* q = &sBl[wn + nt * 16 + lr][8 * lh];
      bH[nt] = frag16(*(const v8h*)p, *(const v8h*)(p + 16));
      bL[nt] = frag16(*(const v8h*)q, *(const v8h*)(q + 16));
    }
    #pragma unroll
    for (int mt = 0; mt < 2; ++mt)
      #pragma unroll
      for (int nt = 0; nt < 2; ++nt) {
        acc[mt][nt] = __builtin_amdgcn_wmma_f32_16x16x32_f16(
            false, aH[mt], false, bH[nt], (short)0, acc[mt][nt], false, false);
        acc[mt][nt] = __builtin_amdgcn_wmma_f32_16x16x32_f16(
            false, aH[mt], false, bL[nt], (short)0, acc[mt][nt], false, false);
        acc[mt][nt] = __builtin_amdgcn_wmma_f32_16x16x32_f16(
            false, aL[mt], false, bH[nt], (short)0, acc[mt][nt], false, false);
      }
    __syncthreads();
  }
}

// ---------------------------------------------------------------------------
// GEMM1: u[16384,256] = x·W_inᵀ + b_in   (K = 1024)
// ---------------------------------------------------------------------------
__global__ __launch_bounds__(256) void k_gemm_in(
    const _Float16* __restrict__ xh, const _Float16* __restrict__ xl,
    const _Float16* __restrict__ wh, const _Float16* __restrict__ wl,
    const float* __restrict__ b_in, float* __restrict__ u) {
  __shared__ _Float16 sAh[128][40], sAl[128][40], sBh[64][40], sBl[64][40];
  v8f acc[2][2];
  #pragma unroll
  for (int mt = 0; mt < 2; ++mt)
    #pragma unroll
    for (int nt = 0; nt < 2; ++nt) acc[mt][nt] = {};

  const int m0 = blockIdx.y * 128, n0 = blockIdx.x * 64;
  gemm_mainloop(xh, xl, DIM, wh, wl, DIM, DIM, m0, n0, sAh, sAl, sBh, sBl, acc);

  const int tid = threadIdx.x, wave = tid >> 5, lane = tid & 31;
  const int wm = (wave >> 1) * 32, wn = (wave & 1) * 32;
  const int en = lane & 15, eh = lane >> 4;
  #pragma unroll
  for (int mt = 0; mt < 2; ++mt)
    #pragma unroll
    for (int nt = 0; nt < 2; ++nt) {
      int gn = n0 + wn + nt * 16 + en;
      float bb = b_in[gn];
      #pragma unroll
      for (int v = 0; v < 8; ++v) {
        int gm = m0 + wm + mt * 16 + eh * 8 + v;
        u[(size_t)gm * NSTATE + gn] = acc[mt][nt][v] + bb;
      }
    }
}

// ---------------------------------------------------------------------------
// Scan phase A: per-chunk local carries.  grid = B*NCH blocks, 256 threads(n)
// ---------------------------------------------------------------------------
__global__ __launch_bounds__(256) void k_scan_carry(
    const float* __restrict__ u, const float* __restrict__ log_decay,
    float* __restrict__ carry) {
  const int blk = blockIdx.x;
  const int b = blk >> 4, ch = blk & 15;
  const int n = threadIdx.x;
  const float dec = 1.0f / (1.0f + expf(-log_decay[n]));
  size_t base = ((size_t)b * SEQ + ch * CHUNK) * NSTATE + n;
  float st = 0.0f;
  for (int t = 0; t < CHUNK; ++t) st = dec * st + u[base + (size_t)t * NSTATE];
  carry[(size_t)(b * NCH + ch) * NSTATE + n] = st;
}

// ---------------------------------------------------------------------------
// Scan phase B: serial prefix over 16 chunk carries. 1 block, 1024 threads.
// prefix[k] = global state at end of chunk k.
// ---------------------------------------------------------------------------
__global__ __launch_bounds__(1024) void k_scan_prefix(
    const float* __restrict__ carry, const float* __restrict__ log_decay,
    float* __restrict__ prefix) {
  const int t = threadIdx.x;
  const int b = t >> 8, n = t & 255;
  const float dec = 1.0f / (1.0f + expf(-log_decay[n]));
  float dC = dec;
  #pragma unroll
  for (int i = 0; i < 8; ++i) dC *= dC;  // dec^256
  float run = carry[(size_t)(b * NCH) * NSTATE + n];
  prefix[(size_t)(b * NCH) * NSTATE + n] = run;
  for (int k = 1; k < NCH; ++k) {
    run = dC * run + carry[(size_t)(b * NCH + k) * NSTATE + n];
    prefix[(size_t)(b * NCH + k) * NSTATE + n] = run;
  }
}

// ---------------------------------------------------------------------------
// Scan phase C: re-scan each chunk seeded with the previous chunk's global
// carry (exact), emit states as hi/lo f16.
// ---------------------------------------------------------------------------
__global__ __launch_bounds__(256) void k_scan_apply(
    const float* __restrict__ u, const float* __restrict__ log_decay,
    const float* __restrict__ prefix,
    _Float16* __restrict__ sh, _Float16* __restrict__ sl) {
  const int blk = blockIdx.x;
  const int b = blk >> 4, ch = blk & 15;
  const int n = threadIdx.x;
  const float dec = 1.0f / (1.0f + expf(-log_decay[n]));
  float st = (ch == 0) ? 0.0f
                       : prefix[(size_t)(b * NCH + ch - 1) * NSTATE + n];
  size_t base = ((size_t)b * SEQ + ch * CHUNK) * NSTATE + n;
  for (int t = 0; t < CHUNK; ++t) {
    size_t idx = base + (size_t)t * NSTATE;
    st = dec * st + u[idx];
    HL s = split16(st);
    sh[idx] = s.h;
    sl[idx] = s.l;
  }
}

// ---------------------------------------------------------------------------
// GEMM2 (fused): out = tokens + states·W_outᵀ (K=256) + x·W_skipᵀ (K=1024)
//                + b_out + b_skip
// ---------------------------------------------------------------------------
__global__ __launch_bounds__(256) void k_gemm_out(
    const _Float16* __restrict__ sh, const _Float16* __restrict__ sl,
    const _Float16* __restrict__ woh, const _Float16* __restrict__ wol,
    const _Float16* __restrict__ xh, const _Float16* __restrict__ xl,
    const _Float16* __restrict__ wsh, const _Float16* __restrict__ wsl,
    const float* __restrict__ tokens,
    const float* __restrict__ b_out, const float* __restrict__ b_skip,
    float* __restrict__ out) {
  __shared__ _Float16 sAh[128][40], sAl[128][40], sBh[64][40], sBl[64][40];
  v8f acc[2][2];
  #pragma unroll
  for (int mt = 0; mt < 2; ++mt)
    #pragma unroll
    for (int nt = 0; nt < 2; ++nt) acc[mt][nt] = {};

  const int m0 = blockIdx.y * 128, n0 = blockIdx.x * 64;
  // y = states · W_outᵀ  (A: [16384,256], B: W_out rows d over n, ld=256)
  gemm_mainloop(sh, sl, NSTATE, woh, wol, NSTATE, NSTATE, m0, n0,
                sAh, sAl, sBh, sBl, acc);
  // y += x · W_skipᵀ     (A: [16384,1024], B: W_skip rows e over d, ld=1024)
  gemm_mainloop(xh, xl, DIM, wsh, wsl, DIM, DIM, m0, n0,
                sAh, sAl, sBh, sBl, acc);

  const int tid = threadIdx.x, wave = tid >> 5, lane = tid & 31;
  const int wm = (wave >> 1) * 32, wn = (wave & 1) * 32;
  const int en = lane & 15, eh = lane >> 4;
  #pragma unroll
  for (int mt = 0; mt < 2; ++mt)
    #pragma unroll
    for (int nt = 0; nt < 2; ++nt) {
      int gn = n0 + wn + nt * 16 + en;
      float bb = b_out[gn] + b_skip[gn];
      #pragma unroll
      for (int v = 0; v < 8; ++v) {
        int gm = m0 + wm + mt * 16 + eh * 8 + v;
        size_t idx = (size_t)gm * DIM + gn;
        out[idx] = tokens[idx] + acc[mt][nt][v] + bb;
      }
    }
}

// ---------------------------------------------------------------------------
// Host-side launch
// ---------------------------------------------------------------------------
extern "C" void kernel_launch(void* const* d_in, const int* in_sizes, int n_in,
                              void* d_out, int out_size, void* d_ws,
                              size_t ws_size, hipStream_t stream) {
  const float* tokens    = (const float*)d_in[0];
  const float* norm_w    = (const float*)d_in[1];
  const float* W_in      = (const float*)d_in[2];
  const float* b_in      = (const float*)d_in[3];
  const float* W_out     = (const float*)d_in[4];
  const float* b_out     = (const float*)d_in[5];
  const float* W_skip    = (const float*)d_in[6];
  const float* b_skip    = (const float*)d_in[7];
  const float* log_decay = (const float*)d_in[8];
  float* out = (float*)d_out;
  char*  ws  = (char*)d_ws;

  // Workspace layout (all offsets 256B-aligned); total ≈ 102.1 MB
  const size_t SZ_X   = (size_t)MTOK * DIM * sizeof(_Float16);     // 32 MB
  const size_t SZ_WIN = (size_t)NSTATE * DIM * sizeof(_Float16);   // 512 KB
  const size_t SZ_WO  = (size_t)DIM * NSTATE * sizeof(_Float16);   // 512 KB
  const size_t SZ_WSK = (size_t)DIM * DIM * sizeof(_Float16);      // 2 MB
  const size_t SZ_U   = (size_t)MTOK * NSTATE * sizeof(float);     // 16 MB
  const size_t SZ_CAR = (size_t)BATCH * NCH * NSTATE * sizeof(float); // 64 KB
  const size_t SZ_S   = (size_t)MTOK * NSTATE * sizeof(_Float16);  // 8 MB

  size_t off = 0;
  _Float16* x_hi  = (_Float16*)(ws + off); off += SZ_X;
  _Float16* x_lo  = (_Float16*)(ws + off); off += SZ_X;
  _Float16* win_h = (_Float16*)(ws + off); off += SZ_WIN;
  _Float16* win_l = (_Float16*)(ws + off); off += SZ_WIN;
  _Float16* wo_h  = (_Float16*)(ws + off); off += SZ_WO;
  _Float16* wo_l  = (_Float16*)(ws + off); off += SZ_WO;
  _Float16* wsk_h = (_Float16*)(ws + off); off += SZ_WSK;
  _Float16* wsk_l = (_Float16*)(ws + off); off += SZ_WSK;
  float*    u     = (float*)   (ws + off); off += SZ_U;
  float*    carry = (float*)   (ws + off); off += SZ_CAR;
  float*    prefx = (float*)   (ws + off); off += SZ_CAR;
  _Float16* s_hi  = (_Float16*)(ws + off); off += SZ_S;
  _Float16* s_lo  = (_Float16*)(ws + off); off += SZ_S;

  // 1) weight splits
  {
    int n4 = NSTATE * DIM / 4;
    k_convert_split<<<(n4 + 255) / 256, 256, 0, stream>>>(W_in, win_h, win_l, n4);
    k_convert_split<<<(n4 + 255) / 256, 256, 0, stream>>>(W_out, wo_h, wo_l, n4);
    int m4 = DIM * DIM / 4;
    k_convert_split<<<(m4 + 255) / 256, 256, 0, stream>>>(W_skip, wsk_h, wsk_l, m4);
  }
  // 2) RMSNorm + split
  k_rmsnorm_split<<<MTOK, 256, 0, stream>>>(tokens, norm_w, x_hi, x_lo);
  // 3) u = x·W_inᵀ + b_in
  {
    dim3 grid(NSTATE / 64, MTOK / 128);
    k_gemm_in<<<grid, 256, 0, stream>>>(x_hi, x_lo, win_h, win_l, b_in, u);
  }
  // 4) chunked linear-recurrence scan (exact)
  k_scan_carry<<<BATCH * NCH, 256, 0, stream>>>(u, log_decay, carry);
  k_scan_prefix<<<1, BATCH * NSTATE, 0, stream>>>(carry, log_decay, prefx);
  k_scan_apply<<<BATCH * NCH, 256, 0, stream>>>(u, log_decay, prefx, s_hi, s_lo);
  // 5) fused output GEMM + residual + biases
  {
    dim3 grid(DIM / 64, MTOK / 128);
    k_gemm_out<<<grid, 256, 0, stream>>>(s_hi, s_lo, wo_h, wo_l,
                                         x_hi, x_lo, wsk_h, wsk_l,
                                         tokens, b_out, b_skip, out);
  }
}
